// StackedLSTMCell_2018634629836
// MI455X (gfx1250) — compile-verified, matrix operand order checked by reading, not run on hardware
//
#include <hip/hip_runtime.h>
#include <hip/hip_bf16.h>

typedef __bf16 bf16;
typedef bf16  v4bf  __attribute__((ext_vector_type(4)));
typedef bf16  v8bf  __attribute__((ext_vector_type(8)));
typedef bf16  v16bf __attribute__((ext_vector_type(16)));
typedef float v4f   __attribute__((ext_vector_type(4)));
typedef float v8f   __attribute__((ext_vector_type(8)));
typedef int   v4i   __attribute__((ext_vector_type(4)));

constexpr int HID    = 256;     // S
constexpr int NBATCH = 32768;   // B
constexpr int NLAYER = 4;       // L
constexpr int BM     = 256;     // block M tile
constexpr int KB     = 32;      // K per step
constexpr int LDA    = 40;      // LDS row stride (bf16 elems) = 80B
constexpr int NSTEP  = 2 * (HID / KB);  // 16

#if __has_builtin(__builtin_amdgcn_global_load_async_to_lds_b128) && \
    __has_builtin(__builtin_amdgcn_s_wait_asynccnt)
#define HAVE_ASYNC 1
#else
#define HAVE_ASYNC 0
#endif

__device__ __forceinline__ float sigmoidf_(float x) {
  return 1.0f / (1.0f + __expf(-x));
}
__device__ __forceinline__ float tanhf_(float x) {
  x = fminf(fmaxf(x, -15.0f), 15.0f);
  float e = __expf(2.0f * x);
  return (e - 1.0f) / (e + 1.0f);
}

// A/B fragment per CDNA5 16-bit layout: lane<16 K={0..7,16..23}, lane>=16 +8.
__device__ __forceinline__ v16bf load_frag(const bf16* base, int row, int khalf) {
  const v8bf* p0 = (const v8bf*)(base + row * LDA + khalf * 8);
  const v8bf* p1 = (const v8bf*)(base + row * LDA + khalf * 8 + 16);
  v8bf lo = *p0, hi = *p1;
  v16bf r;
#pragma unroll
  for (int i = 0; i < 8; ++i) { r[i] = lo[i]; r[i + 8] = hi[i]; }
  return r;
}

// Shared LSTM-cell epilogue. C/D layout: VGPR r -> rows r / r+8, col N = lane&15.
__device__ __forceinline__ void cell_epilogue(
    v8f (&acc)[2][4], const float* __restrict__ cprev,
    const float* __restrict__ bih, const float* __restrict__ bhh,
    float* __restrict__ hout, float* __restrict__ cout, bf16* __restrict__ xnext,
    int bm0, int sn0, int wave, int lane) {
  const int lrow = lane & 15, khalf = lane >> 4;
  const int scol = sn0 + lrow;
  float biasv[4];
#pragma unroll
  for (int g = 0; g < 4; ++g)
    biasv[g] = bih[g * HID + scol] + bhh[g * HID + scol];
#pragma unroll
  for (int mt = 0; mt < 2; ++mt) {
#pragma unroll
    for (int r = 0; r < 8; ++r) {
      int grow   = bm0 + wave * 32 + mt * 16 + r + khalf * 8;
      size_t off = (size_t)grow * HID + scol;
      float iv = sigmoidf_(acc[mt][0][r] + biasv[0]);
      float fv = sigmoidf_(acc[mt][1][r] + biasv[1]);
      float gv = tanhf_  (acc[mt][2][r] + biasv[2]);
      float ov = sigmoidf_(acc[mt][3][r] + biasv[3]);
      float cn = fv * cprev[off] + iv * gv;
      float hn = ov * tanhf_(cn);
      cout[off] = cn;
      hout[off] = hn;
      if (xnext) xnext[off] = (bf16)hn;
    }
  }
}

// ---------------------------------------------------------------------------
// f32 -> bf16 conversion pre-pass (vectorized x4)
// ---------------------------------------------------------------------------
__global__ __launch_bounds__(256)
void cvt_f32_bf16_kernel(const float* __restrict__ src, bf16* __restrict__ dst,
                         int n4) {
  int i = blockIdx.x * 256 + threadIdx.x;
  if (i < n4) {
    v4f a = ((const v4f*)src)[i];
    v4bf b;
#pragma unroll
    for (int j = 0; j < 4; ++j) b[j] = (bf16)a[j];
    ((v4bf*)dst)[i] = b;
  }
}

#if HAVE_ASYNC
// ---------------------------------------------------------------------------
// Fast path: bf16 operands staged with global_load_async_to_lds_b128 (ASYNCcnt)
// ---------------------------------------------------------------------------
typedef __attribute__((address_space(1))) v4i* gv4_t;
typedef __attribute__((address_space(3))) v4i* lv4_t;

__device__ __forceinline__ void async_g2l_b128(const void* g, void* l) {
  __builtin_amdgcn_global_load_async_to_lds_b128((gv4_t)g, (lv4_t)l, 0, 0);
}

__global__ __launch_bounds__(256)
void lstm_layer_async(const bf16* __restrict__ xbf, const bf16* __restrict__ hbf,
                      const float* __restrict__ cprev,
                      const bf16* __restrict__ Wihb, const bf16* __restrict__ Whhb,
                      const float* __restrict__ bih, const float* __restrict__ bhh,
                      float* __restrict__ hout, float* __restrict__ cout,
                      bf16* __restrict__ xnext) {
  __shared__ __align__(16) bf16 As[2][BM * LDA];
  __shared__ __align__(16) bf16 Bs[2][64 * LDA];

  const int tid   = threadIdx.x;
  const int lane  = tid & 31;
  const int wave  = tid >> 5;
  const int khalf = lane >> 4;
  const int lrow  = lane & 15;
  const int bm0   = blockIdx.x * BM;
  const int sn0   = blockIdx.y * 16;

  // Per-thread async copy assignments, constant across K-steps.
  // A tile: 256 rows x 32 bf16 = 1024 x 16B chunks -> 4 per thread.
  int aoffG[4], aoffL[4];
#pragma unroll
  for (int i = 0; i < 4; ++i) {
    int c    = tid + i * 256;
    int arow = c >> 2;
    int kc8  = (c & 3) * 8;
    aoffG[i] = (bm0 + arow) * HID + kc8;
    aoffL[i] = arow * LDA + kc8;
  }
  // B tile: 64 rows x 32 bf16 = 256 x 16B chunks -> 1 per thread.
  const int brow  = tid >> 2;
  const int bkc8  = (tid & 3) * 8;
  const int boffG = ((brow >> 4) * HID + sn0 + (brow & 15)) * HID + bkc8;
  const int boffL = brow * LDA + bkc8;

  v8f acc[2][4];
#pragma unroll
  for (int mt = 0; mt < 2; ++mt)
#pragma unroll
    for (int g = 0; g < 4; ++g)
      acc[mt][g] = (v8f){0.f, 0.f, 0.f, 0.f, 0.f, 0.f, 0.f, 0.f};

  // Prologue: stage step 0.
  {
    const bf16* asrc = xbf;
    const bf16* bsrc = Wihb;
#pragma unroll
    for (int i = 0; i < 4; ++i)
      async_g2l_b128(asrc + aoffG[i], &As[0][aoffL[i]]);
    async_g2l_b128(bsrc + boffG, &Bs[0][boffL]);
    __builtin_amdgcn_s_wait_asynccnt(0);
  }
  __syncthreads();

#pragma unroll 2
  for (int step = 0; step < NSTEP; ++step) {
    const int  cur  = step & 1;
    const bool more = (step + 1) < NSTEP;

    if (more) {
      const int   s1    = step + 1;
      const bf16* asrc  = (s1 < NSTEP / 2) ? xbf : hbf;
      const bf16* bsrc  = (s1 < NSTEP / 2) ? Wihb : Whhb;
      const int   kbase = (s1 & (NSTEP / 2 - 1)) * KB;
#pragma unroll
      for (int i = 0; i < 4; ++i)
        async_g2l_b128(asrc + kbase + aoffG[i], &As[cur ^ 1][aoffL[i]]);
      async_g2l_b128(bsrc + kbase + boffG, &Bs[cur ^ 1][boffL]);
    }

    v16bf bfr[4];
#pragma unroll
    for (int g = 0; g < 4; ++g)
      bfr[g] = load_frag(Bs[cur], g * 16 + lrow, khalf);
#pragma unroll
    for (int mt = 0; mt < 2; ++mt) {
      v16bf af = load_frag(As[cur], wave * 32 + mt * 16 + lrow, khalf);
#pragma unroll
      for (int g = 0; g < 4; ++g)
        acc[mt][g] = __builtin_amdgcn_wmma_f32_16x16x32_bf16(
            false, af, false, bfr[g], (short)0, acc[mt][g], false, false);
    }

    if (more) __builtin_amdgcn_s_wait_asynccnt(0);
    __syncthreads();
  }

  cell_epilogue(acc, cprev, bih, bhh, hout, cout, xnext, bm0, sn0, wave, lane);
}
#endif  // HAVE_ASYNC

// ---------------------------------------------------------------------------
// Fallback path: f32 global loads + in-kernel cvt + ds_store (round-1 scheme)
// ---------------------------------------------------------------------------
__device__ __forceinline__ void fetch_tiles(int step,
    const float* __restrict__ x, const float* __restrict__ h,
    const float* __restrict__ Wih, const float* __restrict__ Whh,
    int bm0, int sn0, int tid, v4f* aR, v4f* bR) {
  const float* asrc = (step < NSTEP / 2) ? x : h;
  const float* bsrc = (step < NSTEP / 2) ? Wih : Whh;
  const int kbase = (step & (NSTEP / 2 - 1)) * KB;
#pragma unroll
  for (int i = 0; i < 8; ++i) {
    int idx = tid + i * 256, arow = idx >> 3, kc = (idx & 7) << 2;
    aR[i] = *(const v4f*)(asrc + (size_t)(bm0 + arow) * HID + kbase + kc);
  }
#pragma unroll
  for (int i = 0; i < 2; ++i) {
    int idx = tid + i * 256, brow = idx >> 3, kc = (idx & 7) << 2;
    int wrow = (brow >> 4) * HID + sn0 + (brow & 15);
    bR[i] = *(const v4f*)(bsrc + (size_t)wrow * HID + kbase + kc);
  }
}

__device__ __forceinline__ void stage_tiles(const v4f* aR, const v4f* bR,
                                            bf16* Asd, bf16* Bsd, int tid) {
#pragma unroll
  for (int i = 0; i < 8; ++i) {
    int idx = tid + i * 256, arow = idx >> 3, kc = (idx & 7) << 2;
    v4bf v;
#pragma unroll
    for (int j = 0; j < 4; ++j) v[j] = (bf16)aR[i][j];
    *(v4bf*)(Asd + arow * LDA + kc) = v;
  }
#pragma unroll
  for (int i = 0; i < 2; ++i) {
    int idx = tid + i * 256, brow = idx >> 3, kc = (idx & 7) << 2;
    v4bf v;
#pragma unroll
    for (int j = 0; j < 4; ++j) v[j] = (bf16)bR[i][j];
    *(v4bf*)(Bsd + brow * LDA + kc) = v;
  }
}

__global__ __launch_bounds__(256)
void lstm_layer_fallback(const float* __restrict__ x,
                         const float* __restrict__ hprev,
                         const float* __restrict__ cprev,
                         const float* __restrict__ Wih,
                         const float* __restrict__ Whh,
                         const float* __restrict__ bih,
                         const float* __restrict__ bhh,
                         float* __restrict__ hout,
                         float* __restrict__ cout) {
  __shared__ __align__(16) bf16 As[2][BM * LDA];
  __shared__ __align__(16) bf16 Bs[2][64 * LDA];

  const int tid = threadIdx.x, lane = tid & 31, wave = tid >> 5;
  const int khalf = lane >> 4, lrow = lane & 15;
  const int bm0 = blockIdx.x * BM, sn0 = blockIdx.y * 16;

  v8f acc[2][4];
#pragma unroll
  for (int mt = 0; mt < 2; ++mt)
#pragma unroll
    for (int g = 0; g < 4; ++g)
      acc[mt][g] = (v8f){0.f, 0.f, 0.f, 0.f, 0.f, 0.f, 0.f, 0.f};

  {
    v4f aR[8], bR[2];
    fetch_tiles(0, x, hprev, Wih, Whh, bm0, sn0, tid, aR, bR);
    stage_tiles(aR, bR, As[0], Bs[0], tid);
  }
  __syncthreads();

#pragma unroll 2
  for (int step = 0; step < NSTEP; ++step) {
    const int  cur  = step & 1;
    const bool more = (step + 1) < NSTEP;
    v4f aN[8], bN[2];
    if (more) fetch_tiles(step + 1, x, hprev, Wih, Whh, bm0, sn0, tid, aN, bN);

    v16bf bfr[4];
#pragma unroll
    for (int g = 0; g < 4; ++g)
      bfr[g] = load_frag(Bs[cur], g * 16 + lrow, khalf);
#pragma unroll
    for (int mt = 0; mt < 2; ++mt) {
      v16bf af = load_frag(As[cur], wave * 32 + mt * 16 + lrow, khalf);
#pragma unroll
      for (int g = 0; g < 4; ++g)
        acc[mt][g] = __builtin_amdgcn_wmma_f32_16x16x32_bf16(
            false, af, false, bfr[g], (short)0, acc[mt][g], false, false);
    }

    if (more) stage_tiles(aN, bN, As[cur ^ 1], Bs[cur ^ 1], tid);
    __syncthreads();
  }

  cell_epilogue(acc, cprev, bih, bhh, hout, cout, nullptr, bm0, sn0, wave, lane);
}

// ---------------------------------------------------------------------------
extern "C" void kernel_launch(void* const* d_in, const int* in_sizes, int n_in,
                              void* d_out, int out_size, void* d_ws, size_t ws_size,
                              hipStream_t stream) {
  const float* x0  = (const float*)d_in[0];   // [B,S]
  const float* ph  = (const float*)d_in[1];   // [L,B,S]
  const float* pc  = (const float*)d_in[2];   // [L,B,S]
  const float* Wih = (const float*)d_in[3];   // [L,4S,S]
  const float* Whh = (const float*)d_in[4];   // [L,4S,S]
  const float* bih = (const float*)d_in[5];   // [L,4S]
  const float* bhh = (const float*)d_in[6];   // [L,4S]
  float* out = (float*)d_out;                 // next_h [L,B,S] ++ next_c [L,B,S]

  const size_t BS  = (size_t)NBATCH * HID;
  const size_t GS  = (size_t)4 * HID * HID;       // per-layer weight elems
  const size_t LGS = (size_t)NLAYER * GS;         // all-layer weight elems
  const size_t GB  = (size_t)4 * HID;

  dim3 grid(NBATCH / BM, HID / 16);
  dim3 block(256);

#if HAVE_ASYNC
  const size_t need = (3 * BS + 2 * LGS) * sizeof(bf16);
  if (ws_size >= need) {
    bf16* xbf0 = (bf16*)d_ws;
    bf16* xbf1 = xbf0 + BS;
    bf16* hbf  = xbf1 + BS;
    bf16* wihb = hbf + BS;
    bf16* whhb = wihb + LGS;

    // One-time conversions: x and all weights.
    cvt_f32_bf16_kernel<<<(int)(BS / 4 + 255) / 256, 256, 0, stream>>>(x0, xbf0, (int)(BS / 4));
    cvt_f32_bf16_kernel<<<(int)(LGS / 4 + 255) / 256, 256, 0, stream>>>(Wih, wihb, (int)(LGS / 4));
    cvt_f32_bf16_kernel<<<(int)(LGS / 4 + 255) / 256, 256, 0, stream>>>(Whh, whhb, (int)(LGS / 4));

    for (int l = 0; l < NLAYER; ++l) {
      // Per-layer prev_h conversion (stream-ordered with the GEMM below).
      cvt_f32_bf16_kernel<<<(int)(BS / 4 + 255) / 256, 256, 0, stream>>>(
          ph + l * BS, hbf, (int)(BS / 4));
      const bf16* xin   = (l & 1) ? xbf1 : xbf0;
      bf16*       xnext = (l & 1) ? xbf0 : xbf1;
      lstm_layer_async<<<grid, block, 0, stream>>>(
          xin, hbf, pc + l * BS,
          wihb + l * GS, whhb + l * GS,
          bih + l * GB, bhh + l * GB,
          out + l * BS, out + (size_t)NLAYER * BS + l * BS,
          xnext);
    }
    return;
  }
#endif

  // Fallback: convert-in-kernel path (no scratch requirements).
  for (int l = 0; l < NLAYER; ++l) {
    const float* xin = (l == 0) ? x0 : (out + (size_t)(l - 1) * BS);
    lstm_layer_fallback<<<grid, block, 0, stream>>>(
        xin, ph + l * BS, pc + l * BS,
        Wih + l * GS, Whh + l * GS,
        bih + l * GB, bhh + l * GB,
        out + l * BS, out + (size_t)NLAYER * BS + l * BS);
  }
}
